// GPTWithCoT_78615081386342
// MI455X (gfx1250) — matrix operand order repaired and checked by reading, not compiled, not run on hardware
//
#include <hip/hip_runtime.h>
#include <hip/hip_bf16.h>
#include <math.h>

// ---------------- problem constants ----------------
#define Bv    8
#define Tv    512
#define Dv    768
#define NHv   12
#define NLv   12
#define LKSv  10
#define SDv   50
#define ADv   20
#define HSv   64
#define TTv   522           // T + LKS
#define TTP   528           // padded to multiple of 16
#define MROWS (Bv * TTP)    // 4224 padded token rows

typedef __attribute__((ext_vector_type(16))) _Float16 v16h;
typedef __attribute__((ext_vector_type(8)))  _Float16 v8h;
typedef __attribute__((ext_vector_type(8)))  float    v8f;

static inline int cdiv(int a, int b) { return (a + b - 1) / b; }

__device__ __forceinline__ v16h combine16(v8h lo, v8h hi) {
  return __builtin_shufflevector(lo, hi, 0, 1, 2, 3, 4, 5, 6, 7,
                                 8, 9, 10, 11, 12, 13, 14, 15);
}

// ---------------------------------------------------------------------------
// Pack row-major f32 [K,N] into WMMA B-fragment order, f16, zero padded:
//   dst[((kt*nt16+nt)*32 + lane)*16 + e] = B[kt*32 + (lane&16?8:0) + (e<8?e:e+8)][nt*16 + (lane&15)]
// A B-fragment load in the GEMM is then 32 contiguous bytes per lane.
// ---------------------------------------------------------------------------
__global__ void pack_b_kernel(const float* __restrict__ src,
                              _Float16* __restrict__ dst, int K, int N) {
  long idx = (long)blockIdx.x * blockDim.x + threadIdx.x;
  int nt16 = (N + 15) >> 4;
  long total = (long)((K + 31) >> 5) * nt16 * 512;
  if (idx >= total) return;
  long blk = idx >> 9;
  int rpos = (int)(idx & 511);
  int lane = rpos >> 4;
  int e    = rpos & 15;
  int kt   = (int)(blk / nt16);
  int nt   = (int)(blk % nt16);
  int k = kt * 32 + ((lane & 16) ? 8 : 0) + ((e < 8) ? e : e + 8);
  int n = nt * 16 + (lane & 15);
  float v = (k < K && n < N) ? src[(size_t)k * N + n] : 0.f;
  dst[idx] = (_Float16)v;
}

__global__ void convert_h_kernel(const float* __restrict__ src,
                                 _Float16* __restrict__ dst, long n) {
  long i = (long)blockIdx.x * blockDim.x + threadIdx.x;
  if (i < n) dst[i] = (_Float16)src[i];
}

// exact gelu, applied in-place on an f16 buffer (keeps erff out of the GEMM)
__global__ void gelu_kernel(_Float16* __restrict__ buf, long n) {
  long i = (long)blockIdx.x * blockDim.x + threadIdx.x;
  if (i >= n) return;
  float v = (float)buf[i];
  buf[i] = (_Float16)(0.5f * v * (1.f + erff(v * 0.70710678118654752f)));
}

// ---------------------------------------------------------------------------
// Epilogue helper: templated so each instance carries exactly one path.
// OMODE: 0 = f16 out, 1 = f32 out, 2 = f32 out + f32 residual.  ACT: 0/1.
// ---------------------------------------------------------------------------
template <int OMODE, int ACT>
__device__ __forceinline__ void store_one(float v, const float* res,
                                          float* Cf, _Float16* Ch,
                                          size_t off) {
  if (OMODE == 2) v += res[off];
  if (ACT == 1) v = fmaxf(v, 0.f);
  if (OMODE == 0) Ch[off] = (_Float16)v;
  else            Cf[off] = v;
}

// ---------------------------------------------------------------------------
// Fast WMMA GEMM (M%64==0, N%64==0, K%32==0, 16B-aligned f16 A rows).
// One wave computes a 64x64 tile: 16 v_wmma + 16 vector loads per K-step.
// ---------------------------------------------------------------------------
template <int OMODE, int ACT>
__global__ void gemm_fast(const _Float16* __restrict__ A, int lda,
                          const _Float16* __restrict__ Bs,
                          const float* __restrict__ bias,
                          const float* __restrict__ res,
                          float* __restrict__ Cf, _Float16* __restrict__ Ch,
                          int ldc, int M, int N, int K) {
  int lane = threadIdx.x & 31;
  int wave = threadIdx.x >> 5;
  int t64N = N >> 6;
  int tile = blockIdx.x * (blockDim.x >> 5) + wave;
  if (tile >= (M >> 6) * t64N) return;
  int m0 = (tile / t64N) << 6;
  int n0 = (tile % t64N) << 6;
  int l16 = lane & 15;
  int grp = (lane >> 4) & 1;
  int kbaseA = grp ? 8 : 0;

  v8f acc[4][4];
#pragma unroll
  for (int i = 0; i < 4; ++i)
#pragma unroll
    for (int j = 0; j < 4; ++j)
#pragma unroll
      for (int r = 0; r < 8; ++r) acc[i][j][r] = 0.f;

  const _Float16* arow[4];
#pragma unroll
  for (int i = 0; i < 4; ++i)
    arow[i] = A + (size_t)(m0 + i * 16 + l16) * lda + kbaseA;
  const _Float16* bp = Bs + (size_t)(n0 >> 4) * 512 + lane * 16;
  const size_t bstride = (size_t)(N >> 4) * 512;

  for (int k0 = 0; k0 < K; k0 += 32) {
    v16h a[4];
#pragma unroll
    for (int i = 0; i < 4; ++i)
      a[i] = combine16(*(const v8h*)(arow[i] + k0),
                       *(const v8h*)(arow[i] + k0 + 16));
#pragma unroll
    for (int j = 0; j < 4; ++j) {
      v16h bf = *(const v16h*)(bp + (size_t)j * 512);
#pragma unroll
      for (int i = 0; i < 4; ++i)
        acc[i][j] = __builtin_amdgcn_wmma_f32_16x16x32_f16(
            false, a[i], false, bf, (short)0, acc[i][j], false, false);
    }
    bp += bstride;
  }

#pragma unroll
  for (int j = 0; j < 4; ++j) {
    int col = n0 + j * 16 + l16;
    float bvv = bias[col];
#pragma unroll
    for (int i = 0; i < 4; ++i) {
#pragma unroll
      for (int r = 0; r < 8; ++r) {
        size_t off = (size_t)(m0 + i * 16 + r + grp * 8) * ldc + col;
        store_one<OMODE, ACT>(acc[i][j][r] + bvv, res, Cf, Ch, off);
      }
    }
  }
}

// ---------------------------------------------------------------------------
// Edge WMMA GEMM for ragged shapes (K=50, N=50/20, M=80...):
// 16x16 wave tile, clamped scalar A loads (B zero-padding kills K overflow),
// guarded stores.  Small static footprint.
// ---------------------------------------------------------------------------
template <int OMODE, int ACT>
__global__ void gemm_edge(const _Float16* __restrict__ A, int lda,
                          const _Float16* __restrict__ Bs,
                          const float* __restrict__ bias,
                          const float* __restrict__ res,
                          float* __restrict__ Cf, _Float16* __restrict__ Ch,
                          int ldc, int M, int N, int K) {
  int lane = threadIdx.x & 31;
  int wave = threadIdx.x >> 5;
  int t16N = (N + 15) >> 4;
  int tile = blockIdx.x * (blockDim.x >> 5) + wave;
  if (tile >= ((M + 15) >> 4) * t16N) return;
  int m0 = (tile / t16N) << 4;
  int n0 = (tile % t16N) << 4;
  int l16 = lane & 15;
  int grp = (lane >> 4) & 1;
  int kbaseA = grp ? 8 : 0;

  v8f acc;
#pragma unroll
  for (int r = 0; r < 8; ++r) acc[r] = 0.f;

  int arowi = m0 + l16;
  arowi = (arowi < M) ? arowi : (M - 1);
  const _Float16* ar = A + (size_t)arowi * lda;
  const _Float16* bp = Bs + (size_t)(n0 >> 4) * 512 + lane * 16;
  const size_t bstride = (size_t)t16N * 512;

  const int ksteps = (K + 31) >> 5;
  for (int ks = 0; ks < ksteps; ++ks) {
    int k0 = ks * 32;
    v16h a;
#pragma unroll
    for (int e = 0; e < 16; ++e) {
      int k = k0 + kbaseA + ((e < 8) ? e : e + 8);
      k = (k < K) ? k : (K - 1);  // B padding is zero; value irrelevant
      a[e] = ar[k];
    }
    v16h bf = *(const v16h*)bp;
    acc = __builtin_amdgcn_wmma_f32_16x16x32_f16(false, a, false, bf, (short)0,
                                                 acc, false, false);
    bp += bstride;
  }

  int col = n0 + l16;
  if (col >= N) return;
  float bvv = bias[col];
#pragma unroll
  for (int r = 0; r < 8; ++r) {
    int row = m0 + r + grp * 8;
    if (row < M)
      store_one<OMODE, ACT>(acc[r] + bvv, res, Cf, Ch, (size_t)row * ldc + col);
  }
}

// ---------------------------------------------------------------------------
// Per-batch transpose of V: vt[b][col][key] = vh[b*TTP+key][col]  (f16).
// Makes the attention AV B-fragment a contiguous 32-byte load.
// ---------------------------------------------------------------------------
__global__ void transpose_v_kernel(const _Float16* __restrict__ vh,
                                   _Float16* __restrict__ vt) {
  long i = (long)blockIdx.x * blockDim.x + threadIdx.x;
  long total = (long)Bv * Dv * TTP;
  if (i >= total) return;
  int t = (int)(i % TTP);
  int c = (int)((i / TTP) % Dv);
  int b = (int)(i / ((long)TTP * Dv));
  vt[i] = vh[((size_t)b * TTP + t) * Dv + c];
}

// ---------------------------------------------------------------------------
// Flash attention: one wave per (batch, head, 16-query tile). f16 Q/K/Vt.
// 8 v_wmma per 32-key block; all global accesses are 16B vector loads.
// ---------------------------------------------------------------------------
__global__ void attn_kernel(const _Float16* __restrict__ qh,
                            const _Float16* __restrict__ kh,
                            const _Float16* __restrict__ vt,
                            const unsigned char* __restrict__ ksm,
                            _Float16* __restrict__ yh) {
  __shared__ _Float16 pshared[8][16 * 32];

  int lane  = threadIdx.x & 31;
  int wave  = threadIdx.x >> 5;
  int gwave = blockIdx.x * (blockDim.x >> 5) + wave;
  const int MT = TTP / 16;  // 33
  if (gwave >= Bv * NHv * MT) return;

  int mt = gwave % MT;
  int bh = gwave / MT;
  int h  = bh % NHv;
  int b  = bh / NHv;
  int m0 = mt * 16;

  const float scale = 0.125f;  // 1/sqrt(64)
  int grp = (lane & 16) ? 1 : 0;
  int l16 = lane & 15;
  size_t rowbase = (size_t)b * TTP;
  const unsigned char* kmrow = ksm + (size_t)b * NHv * TTv * TTv;  // [b,0,0,:]

  v16h qa0, qa1;
  {
    const _Float16* qr =
        qh + (rowbase + m0 + l16) * (size_t)Dv + h * HSv + (grp ? 8 : 0);
    qa0 = combine16(*(const v8h*)(qr), *(const v8h*)(qr + 16));
    qa1 = combine16(*(const v8h*)(qr + 32), *(const v8h*)(qr + 48));
  }

  v8f o[4];
#pragma unroll
  for (int ns = 0; ns < 4; ++ns)
#pragma unroll
    for (int r = 0; r < 8; ++r) o[ns][r] = 0.f;
  float mi[8], li[8];
#pragma unroll
  for (int r = 0; r < 8; ++r) { mi[r] = -1e30f; li[r] = 0.f; }

  int kend = (m0 < LKSv) ? TTv : ((m0 + 16 < TTv) ? m0 + 16 : TTv);
  _Float16* pw = pshared[wave];
  const _Float16* vtb = vt + ((size_t)b * Dv + h * HSv) * TTP;
  int kk0 = grp ? 16 : 0;

  for (int j0 = 0; j0 < kend; j0 += 32) {
    // ---- S = Q @ K^T (two 16x16 tiles over 32 keys) ----
    v8f s0, s1;
#pragma unroll
    for (int r = 0; r < 8; ++r) { s0[r] = 0.f; s1[r] = 0.f; }
    int key0 = j0 + l16;       key0 = (key0 < TTP) ? key0 : (TTP - 1);
    int key1 = j0 + 16 + l16;  key1 = (key1 < TTP) ? key1 : (TTP - 1);
#pragma unroll
    for (int step = 0; step < 2; ++step) {
      int koff = step * 32 + (grp ? 8 : 0);
      const _Float16* kr0 = kh + (rowbase + key0) * (size_t)Dv + h * HSv + koff;
      const _Float16* kr1 = kh + (rowbase + key1) * (size_t)Dv + h * HSv + koff;
      v16h kf0 = combine16(*(const v8h*)(kr0), *(const v8h*)(kr0 + 16));
      v16h kf1 = combine16(*(const v8h*)(kr1), *(const v8h*)(kr1 + 16));
      v16h qa = step ? qa1 : qa0;
      s0 = __builtin_amdgcn_wmma_f32_16x16x32_f16(false, qa, false, kf0,
                                                  (short)0, s0, false, false);
      s1 = __builtin_amdgcn_wmma_f32_16x16x32_f16(false, qa, false, kf1,
                                                  (short)0, s1, false, false);
    }

    // ---- mask + online softmax ----
    int tk0 = j0 + l16;
    int tk1 = j0 + 16 + l16;
#pragma unroll
    for (int r = 0; r < 8; ++r) {
      int tq = m0 + r + grp * 8;
      bool rq = tq < TTv;
      bool a0 = rq && tk0 < TTv &&
                ((tq < LKSv) ? (kmrow[tk0] == 0) : (tk0 <= tq));
      bool a1 = rq && tk1 < TTv &&
                ((tq < LKSv) ? (kmrow[tk1] == 0) : (tk1 <= tq));
      float v0 = a0 ? s0[r] * scale : -1e30f;
      float v1 = a1 ? s1[r] * scale : -1e30f;
      float mx = fmaxf(v0, v1);
#pragma unroll
      for (int off = 1; off < 16; off <<= 1)
        mx = fmaxf(mx, __shfl_xor(mx, off, 32));
      float mnew  = fmaxf(mi[r], mx);
      float alpha = __expf(mi[r] - mnew);
      float p0 = __expf(v0 - mnew);
      float p1 = __expf(v1 - mnew);
      float rs = p0 + p1;
#pragma unroll
      for (int off = 1; off < 16; off <<= 1) rs += __shfl_xor(rs, off, 32);
      li[r] = li[r] * alpha + rs;
      mi[r] = mnew;
#pragma unroll
      for (int ns = 0; ns < 4; ++ns) o[ns][r] *= alpha;
      int Mrow = r + grp * 8;
      pw[Mrow * 32 + l16]      = (_Float16)p0;
      pw[Mrow * 32 + 16 + l16] = (_Float16)p1;
    }
    asm volatile("s_wait_dscnt 0" ::: "memory");  // DS in-order within wave

    // ---- P in A-fragment layout ----
    v16h pa;
#pragma unroll
    for (int e = 0; e < 16; ++e) {
      int kk = e + (grp ? 16 : 0);
      pa[e] = pw[l16 * 32 + kk];
    }

    // ---- O += P @ V : contiguous key-major loads from vt ----
#pragma unroll
    for (int ns = 0; ns < 4; ++ns) {
      const _Float16* vr = vtb + (size_t)(ns * 16 + l16) * TTP + j0 + kk0;
      v16h vf = combine16(*(const v8h*)(vr), *(const v8h*)(vr + 8));
      o[ns] = __builtin_amdgcn_wmma_f32_16x16x32_f16(false, pa, false, vf,
                                                     (short)0, o[ns], false, false);
    }
  }

#pragma unroll
  for (int ns = 0; ns < 4; ++ns) {
#pragma unroll
    for (int r = 0; r < 8; ++r) {
      int tq = m0 + r + grp * 8;
      float den = li[r];
      float val = (den > 0.f) ? o[ns][r] / den : 0.f;
      yh[(rowbase + tq) * (size_t)Dv + h * HSv + ns * 16 + l16] = (_Float16)val;
    }
  }
}

// ---------------------------------------------------------------------------
// LayerNorm: one block (256 threads) per row of D=768, f32 in -> f16 out.
// ---------------------------------------------------------------------------
__global__ void ln_kernel(const float* __restrict__ x,
                          const float* __restrict__ g,
                          const float* __restrict__ b,
                          _Float16* __restrict__ out) {
  int row = blockIdx.x;
  const float* xr = x + (size_t)row * Dv;
  float s = 0.f, s2 = 0.f;
  for (int d = threadIdx.x; d < Dv; d += blockDim.x) {
    float v = xr[d];
    s += v; s2 += v * v;
  }
#pragma unroll
  for (int off = 16; off >= 1; off >>= 1) {
    s  += __shfl_xor(s,  off, 32);
    s2 += __shfl_xor(s2, off, 32);
  }
  __shared__ float sh[2][8];
  int w = threadIdx.x >> 5;
  if ((threadIdx.x & 31) == 0) { sh[0][w] = s; sh[1][w] = s2; }
  __syncthreads();
  float ts = 0.f, ts2 = 0.f;
#pragma unroll
  for (int i = 0; i < 8; ++i) { ts += sh[0][i]; ts2 += sh[1][i]; }
  float mu  = ts / (float)Dv;
  float var = ts2 / (float)Dv - mu * mu;
  float inv = rsqrtf(var + 1e-5f);
  for (int d = threadIdx.x; d < Dv; d += blockDim.x)
    out[(size_t)row * Dv + d] = (_Float16)((xr[d] - mu) * inv * g[d] + b[d]);
}

// ---------------------------------------------------------------------------
__global__ void assemble_kernel(const float* __restrict__ se,
                                const float* __restrict__ lpe,
                                const float* __restrict__ gpe,
                                const float* __restrict__ kspe,
                                const int* __restrict__ timesteps,
                                float* __restrict__ x) {
  long i = (long)blockIdx.x * blockDim.x + threadIdx.x;
  long total = (long)Bv * TTP * Dv;
  if (i >= total) return;
  int d = (int)(i % Dv);
  int t = (int)((i / Dv) % TTP);
  int b = (int)(i / ((long)Dv * TTP));
  float v;
  if (t < LKSv) {
    v = kspe[t * Dv + d];
  } else if (t < TTv) {
    int tt = t - LKSv;
    v = se[((size_t)(b * Tv + tt)) * Dv + d] +
        gpe[(size_t)timesteps[b] * Dv + d] + lpe[tt * Dv + d];
  } else {
    v = 0.f;  // zero pad rows keep downstream math finite
  }
  x[i] = v;
}

__global__ void gather_f0_kernel(const float* __restrict__ x,
                                 _Float16* __restrict__ f0h) {
  long i = (long)blockIdx.x * blockDim.x + threadIdx.x;
  long total = (long)Bv * LKSv * Dv;
  if (i >= total) return;
  int d = (int)(i % Dv);
  int t = (int)((i / Dv) % LKSv);
  int b = (int)(i / ((long)Dv * LKSv));
  f0h[i] = (_Float16)x[((size_t)(b * TTP + t)) * Dv + d];
}

__global__ void scatter_act_kernel(const float* __restrict__ apo,
                                   float* __restrict__ out) {
  long i = (long)blockIdx.x * blockDim.x + threadIdx.x;
  long total = (long)Bv * Tv * ADv;
  if (i >= total) return;
  int a = (int)(i % ADv);
  int t = (int)((i / ADv) % Tv);
  int b = (int)(i / ((long)ADv * Tv));
  out[i] = apo[((size_t)(b * TTP + LKSv + t)) * ADv + a];
}

// ---------------------------------------------------------------------------
extern "C" void kernel_launch(void* const* d_in, const int* in_sizes, int n_in,
                              void* d_out, int out_size, void* d_ws,
                              size_t ws_size, hipStream_t stream) {
  const float* states  = (const float*)d_in[0];
  const int* timesteps = (const int*)d_in[1];
  const unsigned char* ksm = (const unsigned char*)d_in[2];
  const float* lpe  = (const float*)d_in[3];
  const float* gpe  = (const float*)d_in[4];
  const float* kspe = (const float*)d_in[5];
  const float* se_w1 = (const float*)d_in[6];
  const float* se_b1 = (const float*)d_in[7];
  const float* se_w2 = (const float*)d_in[8];
  const float* se_b2 = (const float*)d_in[9];
  const float* ln1_g = (const float*)d_in[10];
  const float* ln1_b = (const float*)d_in[11];
  const float* ln2_g = (const float*)d_in[12];
  const float* ln2_b = (const float*)d_in[13];
  const float* wq = (const float*)d_in[14];
  const float* bq = (const float*)d_in[15];
  const float* wk = (const float*)d_in[16];
  const float* bk = (const float*)d_in[17];
  const float* wv = (const float*)d_in[18];
  const float* bv = (const float*)d_in[19];
  const float* wo = (const float*)d_in[20];
  const float* bo = (const float*)d_in[21];
  const float* w_fc = (const float*)d_in[22];
  const float* b_fc = (const float*)d_in[23];
  const float* w_pr = (const float*)d_in[24];
  const float* b_pr = (const float*)d_in[25];
  const float* lnf_g = (const float*)d_in[26];
  const float* lnf_b = (const float*)d_in[27];
  const float* ap_w1 = (const float*)d_in[28];
  const float* ap_b1 = (const float*)d_in[29];
  const float* ap_w2 = (const float*)d_in[30];
  const float* ap_b2 = (const float*)d_in[31];
  const float* ap_w3 = (const float*)d_in[32];
  const float* ap_b3 = (const float*)d_in[33];
  const float* ksp_w1 = (const float*)d_in[34];
  const float* ksp_b1 = (const float*)d_in[35];
  const float* ksp_w2 = (const float*)d_in[36];
  const float* ksp_b2 = (const float*)d_in[37];

  // ---- workspace carve-out ----
  char* base = (char*)d_ws;
  auto alloc = [&](size_t bytes) {
    char* p = base;
    base += (bytes + 255) & ~(size_t)255;
    return p;
  };
  float*    x    = (float*)alloc((size_t)MROWS * Dv * 4);
  float*    tmpf = (float*)alloc((size_t)MROWS * Dv * 4);   // se out / apo
  _Float16* xlnh = (_Float16*)alloc((size_t)MROWS * Dv * 2);
  _Float16* qh   = (_Float16*)alloc((size_t)MROWS * Dv * 2);
  _Float16* kh   = (_Float16*)alloc((size_t)MROWS * Dv * 2);
  _Float16* vh   = (_Float16*)alloc((size_t)MROWS * Dv * 2);
  _Float16* vtb  = (_Float16*)alloc((size_t)MROWS * Dv * 2 + 256);  // V^T +pad
  _Float16* yh   = (_Float16*)alloc((size_t)MROWS * Dv * 2);
  _Float16* hh   = (_Float16*)alloc((size_t)MROWS * 4 * Dv * 2);
  _Float16* wstg = (_Float16*)alloc((size_t)2359296 * 2 + 256);  // max packed B
  _Float16* sth  = (_Float16*)alloc((size_t)Bv * Tv * SDv * 2 + 256);
  // aliases into hh (lifetimes do not overlap hh's use):
  _Float16* t1h  = hh;
  _Float16* t2h  = hh + (size_t)MROWS * 256;
  _Float16* f0h  = hh;                    // l==0 gap
  _Float16* kshh = hh + (size_t)80 * Dv;  // l==0 gap
  float*    apo  = tmpf;                  // [4224,20], after layers

  float* out_act = (float*)d_out;                    // [8,512,20]
  float* out_ks  = out_act + (size_t)Bv * Tv * ADv;  // [8,10,50]

  auto packB = [&](const float* Bsrc, int K, int N) {
    long pt = (long)((K + 31) / 32) * ((N + 15) / 16) * 512;
    pack_b_kernel<<<cdiv((int)pt, 256), 256, 0, stream>>>(Bsrc, wstg, K, N);
  };
  auto gf = [&](int M, int N) { return cdiv(cdiv(M, 64) * cdiv(N, 64), 8); };
  auto ge = [&](int M, int N) { return cdiv(cdiv(M, 16) * cdiv(N, 16), 8); };

  // ---- state encoder: [4096,50] -> relu 256 -> 768 ----
  {
    long n = (long)Bv * Tv * SDv;
    convert_h_kernel<<<cdiv((int)n, 256), 256, 0, stream>>>(states, sth, n);
  }
  packB(se_w1, SDv, 256);
  gemm_edge<0, 1><<<ge(Bv * Tv, 256), 256, 0, stream>>>(
      sth, SDv, wstg, se_b1, nullptr, nullptr, t1h, 256, Bv * Tv, 256, SDv);
  packB(se_w2, 256, Dv);
  gemm_fast<1, 0><<<gf(Bv * Tv, Dv), 256, 0, stream>>>(
      t1h, 256, wstg, se_b2, nullptr, tmpf, nullptr, Dv, Bv * Tv, Dv, 256);
  {
    long tot = (long)Bv * TTP * Dv;
    assemble_kernel<<<cdiv((int)tot, 256), 256, 0, stream>>>(
        tmpf, lpe, gpe, kspe, timesteps, x);
  }

  const int attn_blocks = cdiv(Bv * NHv * (TTP / 16), 8);
  const long vtN = (long)Bv * Dv * TTP;
  const long hhN = (long)MROWS * 4 * Dv;

  for (int l = 0; l < NLv; ++l) {
    const float* wq_ = wq + (size_t)l * Dv * Dv;
    const float* wk_ = wk + (size_t)l * Dv * Dv;
    const float* wv_ = wv + (size_t)l * Dv * Dv;
    const float* wo_ = wo + (size_t)l * Dv * Dv;
    const float* wf_ = w_fc + (size_t)l * Dv * 4 * Dv;
    const float* wp_ = w_pr + (size_t)l * 4 * Dv * Dv;

    ln_kernel<<<MROWS, 256, 0, stream>>>(x, ln1_g + l * Dv, ln1_b + l * Dv,
                                         xlnh);
    packB(wq_, Dv, Dv);
    gemm_fast<0, 0><<<gf(MROWS, Dv), 256, 0, stream>>>(
        xlnh, Dv, wstg, bq + l * Dv, nullptr, nullptr, qh, Dv, MROWS, Dv, Dv);
    packB(wk_, Dv, Dv);
    gemm_fast<0, 0><<<gf(MROWS, Dv), 256, 0, stream>>>(
        xlnh, Dv, wstg, bk + l * Dv, nullptr, nullptr, kh, Dv, MROWS, Dv, Dv);
    packB(wv_, Dv, Dv);
    gemm_fast<0, 0><<<gf(MROWS, Dv), 256, 0, stream>>>(
        xlnh, Dv, wstg, bv + l * Dv, nullptr, nullptr, vh, Dv, MROWS, Dv, Dv);
    transpose_v_kernel<<<cdiv((int)vtN, 256), 256, 0, stream>>>(vh, vtb);

    attn_kernel<<<attn_blocks, 256, 0, stream>>>(qh, kh, vtb, ksm, yh);

    packB(wo_, Dv, Dv);
    gemm_fast<2, 0><<<gf(MROWS, Dv), 256, 0, stream>>>(
        yh, Dv, wstg, bo + l * Dv, x, x, nullptr, Dv, MROWS, Dv, Dv);

    ln_kernel<<<MROWS, 256, 0, stream>>>(x, ln2_g + l * Dv, ln2_b + l * Dv,
                                         xlnh);
    packB(wf_, Dv, 4 * Dv);
    gemm_fast<0, 0><<<gf(MROWS, 4 * Dv), 256, 0, stream>>>(
        xlnh, Dv, wstg, b_fc + l * 4 * Dv, nullptr, nullptr, hh, 4 * Dv,
        MROWS, 4 * Dv, Dv);
    gelu_kernel<<<cdiv((int)hhN, 256), 256, 0, stream>>>(hh, hhN);
    packB(wp_, 4 * Dv, Dv);
    gemm_fast<2, 0><<<gf(MROWS, Dv), 256, 0, stream>>>(
        hh, 4 * Dv, wstg, b_pr + l * Dv, x, x, nullptr, Dv, MROWS, Dv, 4 * Dv);

    if (l == 0) {  // feats[0][:, :LKS] -> key-state predictor (hh now free)
      long tot = (long)Bv * LKSv * Dv;
      gather_f0_kernel<<<cdiv((int)tot, 256), 256, 0, stream>>>(x, f0h);
      packB(ksp_w1, Dv, 256);
      gemm_edge<0, 1><<<ge(Bv * LKSv, 256), 256, 0, stream>>>(
          f0h, Dv, wstg, ksp_b1, nullptr, nullptr, kshh, 256, Bv * LKSv, 256,
          Dv);
      packB(ksp_w2, 256, SDv);
      gemm_edge<1, 0><<<ge(Bv * LKSv, SDv), 256, 0, stream>>>(
          kshh, 256, wstg, ksp_b2, nullptr, out_ks, nullptr, SDv, Bv * LKSv,
          SDv, 256);
    }
  }

  // ---- final LN + action predictor MLP ----
  ln_kernel<<<MROWS, 256, 0, stream>>>(x, lnf_g, lnf_b, xlnh);
  packB(ap_w1, Dv, 256);
  gemm_fast<0, 1><<<gf(MROWS, 256), 256, 0, stream>>>(
      xlnh, Dv, wstg, ap_b1, nullptr, nullptr, t1h, 256, MROWS, 256, Dv);
  packB(ap_w2, 256, 256);
  gemm_fast<0, 1><<<gf(MROWS, 256), 256, 0, stream>>>(
      t1h, 256, wstg, ap_b2, nullptr, nullptr, t2h, 256, MROWS, 256, 256);
  packB(ap_w3, 256, ADv);
  gemm_edge<1, 0><<<ge(MROWS, ADv), 256, 0, stream>>>(
      t2h, 256, wstg, ap_b3, nullptr, apo, nullptr, ADv, MROWS, ADv, 256);
  {
    long tot = (long)Bv * Tv * ADv;
    scatter_act_kernel<<<cdiv((int)tot, 256), 256, 0, stream>>>(apo, out_act);
  }
}
